// ShortConvolution_74380243632397
// MI455X (gfx1250) — matrix operand order repaired and checked by reading, not compiled, not run on hardware
//
#include <hip/hip_runtime.h>
#include <hip/hip_bf16.h>

// ShortConvolution, fully bf16-WMMA pipeline:
//   pre-pass: x -> bf16, W_exp/W_cmp -> bf16 transposed [n][k]
//   k1: expand GEMM + causal depthwise conv (K=4) -> Y bf16 (ws)
//   k2: compress GEMM -> out f32
// Hot loops use global_load_async_to_lds_b128 (ASYNCcnt) with double-buffered
// LDS so WMMAs overlap the next K-chunk's staging.

#define BATCH 4
#define LSEQ  4096
#define DIM   1024
#define CDIM  2048
#define MTOT  (BATCH * LSEQ)   // 16384

typedef __attribute__((ext_vector_type(16))) __bf16 v16bf;
typedef __attribute__((ext_vector_type(8)))  float  v8f;

union FragU { uint4 q[2]; v16bf v; };

__device__ __forceinline__ unsigned bf16_bits(float f) {
    union { float f; unsigned u; } c; c.f = f;
    unsigned u = c.u;
    return (u + 0x7FFFu + ((u >> 16) & 1u)) >> 16;   // RNE
}
__device__ __forceinline__ unsigned pack_bf16(float a, float b) {
    return bf16_bits(a) | (bf16_bits(b) << 16);
}
__device__ __forceinline__ float bf16_f(unsigned short h) {
    union { unsigned u; float f; } c; c.u = ((unsigned)h) << 16;
    return c.f;
}

// Async 16B copy global -> LDS (no VGPR round-trip, tracked by ASYNCcnt).
__device__ __forceinline__ void async_copy16(void* lds_ptr, const void* gptr) {
    unsigned lds_off = (unsigned)(size_t)lds_ptr;          // generic low 32 = LDS offset
    unsigned long long ga = (unsigned long long)(size_t)gptr;
    asm volatile("global_load_async_to_lds_b128 %0, %1, off"
                 :: "v"(lds_off), "v"(ga) : "memory");
}

// LDS row stride = 40 shorts (80 B = 20 dwords): r*20 mod 64 distinct over 16
// consecutive rows -> conflict-free b128 fragment loads.
#define RS 40

// A fragment (16x32 bf16, MxK): lane m = lane&15; lanes<16: K 0-7,16-23;
// lanes>=16: K 8-15,24-31 (two 16B chunks).
__device__ __forceinline__ v16bf load_frag_a(const unsigned short* sA, int tileRow, int lane) {
    const char* base = (const char*)sA + (size_t)(tileRow * 16 + (lane & 15)) * (RS * 2)
                     + ((lane >> 4) << 4);
    FragU f;
    f.q[0] = *(const uint4*)(base);
    f.q[1] = *(const uint4*)(base + 32);
    return f.v;
}
// B fragment (32x16 bf16, KxN) from K-transposed tile BT[n][k]:
// lane n = lane&15; lanes<16: K 0-7,16-23; lanes>=16: K 8-15,24-31.
__device__ __forceinline__ v16bf load_frag_b(const unsigned short* sB, int nTile, int lane) {
    const char* base = (const char*)sB + (size_t)(nTile * 16 + (lane & 15)) * (RS * 2)
                     + ((lane >> 4) << 4);
    FragU f;
    f.q[0] = *(const uint4*)(base);
    f.q[1] = *(const uint4*)(base + 32);
    return f.v;
}

// ---------------- Pre-pass: x f32 -> bf16 ----------------
__global__ __launch_bounds__(256) void cvt_x_kernel(const float* __restrict__ x,
                                                    unsigned short* __restrict__ Xbf) {
    size_t i = ((size_t)blockIdx.x * 256 + threadIdx.x) * 4;
    float4 v = *(const float4*)(x + i);
    uint2 pk; pk.x = pack_bf16(v.x, v.y); pk.y = pack_bf16(v.z, v.w);
    *(uint2*)(Xbf + i) = pk;
}

// ---------------- Pre-pass: W (KxN f32) -> WT (NxK bf16), tiled transpose ----------------
__global__ __launch_bounds__(256) void transpose_cvt_kernel(const float* __restrict__ src,
                                                            unsigned short* __restrict__ dst,
                                                            int K, int N) {
    __shared__ unsigned short tile[32][33];
    const int kb = blockIdx.x * 32, nb = blockIdx.y * 32;
    const int tx = threadIdx.x & 31, ty = threadIdx.x >> 5;   // 32 x 8
#pragma unroll
    for (int i = 0; i < 32; i += 8) {
        float v = src[(size_t)(kb + ty + i) * N + nb + tx];
        tile[ty + i][tx] = (unsigned short)bf16_bits(v);
    }
    __syncthreads();
#pragma unroll
    for (int i = 0; i < 32; i += 8) {
        dst[(size_t)(nb + ty + i) * K + kb + tx] = tile[tx][ty + i];
    }
}

// ---------------- Kernel 1: expand GEMM + causal depthwise conv ----------------
// Tile: 128 M x 128 N. 9 WMMA M-tiles/wave (tile 0 = 16-row halo, rows m0-16..m0-1).
// Double-buffered async staging: per wave per buffer exactly 5 async b128 issues
// (A padded to 192 rows so issue count is wave-uniform -> s_wait_asynccnt 5 is exact).
#define K1_SA  (192 * RS * 2)   // 15360 B per buffer
#define K1_SB  (128 * RS * 2)   // 10240 B per buffer
#define K1_SMEM (2 * K1_SA + 2 * K1_SB)   // 51200 B (xe spill aliases this)

__device__ __forceinline__ void stage_expand(const unsigned short* __restrict__ Xbf,
                                             const unsigned short* __restrict__ WexpT,
                                             char* sAbuf, char* sBbuf,
                                             int m0, int n0, int k0, int tid) {
    const int row = tid >> 2;       // 0..63
    const int q   = tid & 3;        // 16B quarter of a 64B row
#pragma unroll
    for (int p = 0; p < 3; ++p) {   // A: 192 rows (144 real + clamped pad)
        int r  = p * 64 + row;
        int gm = m0 - 16 + r;
        gm = gm < 0 ? 0 : (gm > MTOT - 1 ? MTOT - 1 : gm);
        async_copy16(sAbuf + (size_t)r * (RS * 2) + q * 16,
                     Xbf + (size_t)gm * DIM + k0 + q * 8);
    }
#pragma unroll
    for (int p = 0; p < 2; ++p) {   // B: 128 rows of WexpT[n][k]
        int r = p * 64 + row;
        async_copy16(sBbuf + (size_t)r * (RS * 2) + q * 16,
                     WexpT + (size_t)(n0 + r) * DIM + k0 + q * 8);
    }
}

__global__ __launch_bounds__(256) void expand_conv_kernel(
    const unsigned short* __restrict__ Xbf, const unsigned short* __restrict__ WexpT,
    const float* __restrict__ bexp, const float* __restrict__ convw,
    const float* __restrict__ convb, unsigned short* __restrict__ Y)
{
    __shared__ __attribute__((aligned(16))) char smem[K1_SMEM];

    const int m0   = blockIdx.x * 128;
    const int n0   = blockIdx.y * 128;
    const int tid  = threadIdx.x;
    const int lane = tid & 31;
    const int wid  = tid >> 5;

    v8f acc[9];
#pragma unroll
    for (int t = 0; t < 9; ++t) acc[t] = (v8f){0, 0, 0, 0, 0, 0, 0, 0};

    stage_expand(Xbf, WexpT, smem, smem + 2 * K1_SA, m0, n0, 0, tid);

    const int NS = DIM / 32;   // 32 K-steps
    for (int s = 0; s < NS; ++s) {
        const int cur = s & 1;
        char* sAcur = smem + (size_t)cur * K1_SA;
        char* sBcur = smem + 2 * K1_SA + (size_t)cur * K1_SB;
        if (s + 1 < NS) {
            char* sAnxt = smem + (size_t)(cur ^ 1) * K1_SA;
            char* sBnxt = smem + 2 * K1_SA + (size_t)(cur ^ 1) * K1_SB;
            stage_expand(Xbf, WexpT, sAnxt, sBnxt, m0, n0, (s + 1) * 32, tid);
            asm volatile("s_wait_asynccnt 5" ::: "memory");   // drain previous buffer only
        } else {
            asm volatile("s_wait_asynccnt 0" ::: "memory");
        }
        __syncthreads();
        v16bf bf = load_frag_b((const unsigned short*)sBcur, wid, lane);
#pragma unroll
        for (int t = 0; t < 9; ++t) {
            v16bf af = load_frag_a((const unsigned short*)sAcur, t, lane);
            acc[t] = __builtin_amdgcn_wmma_f32_16x16x32_bf16(
                false, af, false, bf, (short)0, acc[t], false, false);
        }
        __syncthreads();   // protects buffer cur from next iteration's staging
    }

    // ---- bias + spill xe tile (bf16) to LDS for the conv ----
    unsigned short* sXE = (unsigned short*)smem;   // 144*132*2 = 38016 B, aliases buffers
    {
        const int nloc  = wid * 16 + (lane & 15);
        const float bias = bexp[n0 + nloc];
        const int mbase = (lane >> 4) * 8;
#pragma unroll
        for (int t = 0; t < 9; ++t) {
#pragma unroll
            for (int v = 0; v < 8; ++v) {
                float val = acc[t][v] + bias;
                sXE[(size_t)(t * 16 + mbase + v) * 132 + nloc] = (unsigned short)bf16_bits(val);
            }
        }
    }
    __syncthreads();

    // ---- causal depthwise conv along M, write Y (bf16) ----
    {
        const int cloc = tid & 127;
        const int c    = n0 + cloc;
        const float w0 = convw[0 * CDIM + c];
        const float w1 = convw[1 * CDIM + c];
        const float w2 = convw[2 * CDIM + c];
        const float w3 = convw[3 * CDIM + c];
        const float cb = convb[c];
#pragma unroll 4
        for (int i = 0; i < 64; ++i) {
            const int mi = i * 2 + (tid >> 7);      // 0..127
            const int gm = m0 + mi;
            const int l  = gm & (LSEQ - 1);
            const size_t base = (size_t)(16 + mi) * 132 + cloc;
            float x0  = bf16_f(sXE[base]);
            float xm1 = bf16_f(sXE[base - 132]);
            float xm2 = bf16_f(sXE[base - 2 * 132]);
            float xm3 = bf16_f(sXE[base - 3 * 132]);
            float yv = cb + w3 * x0;                // tap j uses xe[m-3+j]
            if (l >= 1) yv += w2 * xm1;
            if (l >= 2) yv += w1 * xm2;
            if (l >= 3) yv += w0 * xm3;
            Y[(size_t)gm * CDIM + c] = (unsigned short)bf16_bits(yv);
        }
    }
}

// ---------------- Kernel 2: compress GEMM (Y bf16 @ WcmpT) + b_cmp ----------------
#define K2_SA  (128 * RS * 2)   // 10240 B per buffer
#define K2_SMEM (4 * K2_SA)     // 40960 B (2xA + 2xB)

__device__ __forceinline__ void stage_compress(const unsigned short* __restrict__ Y,
                                               const unsigned short* __restrict__ WcmpT,
                                               char* sAbuf, char* sBbuf,
                                               int m0, int n0, int k0, int tid) {
    const int row = tid >> 2;
    const int q   = tid & 3;
#pragma unroll
    for (int p = 0; p < 2; ++p) {
        int r = p * 64 + row;
        async_copy16(sAbuf + (size_t)r * (RS * 2) + q * 16,
                     Y + (size_t)(m0 + r) * CDIM + k0 + q * 8);
        async_copy16(sBbuf + (size_t)r * (RS * 2) + q * 16,
                     WcmpT + (size_t)(n0 + r) * CDIM + k0 + q * 8);
    }
}

__global__ __launch_bounds__(256) void compress_kernel(
    const unsigned short* __restrict__ Y, const unsigned short* __restrict__ WcmpT,
    const float* __restrict__ bcmp, float* __restrict__ out)
{
    __shared__ __attribute__((aligned(16))) char smem[K2_SMEM];

    const int m0   = blockIdx.x * 128;
    const int n0   = blockIdx.y * 128;
    const int tid  = threadIdx.x;
    const int lane = tid & 31;
    const int wid  = tid >> 5;

    v8f acc[8];
#pragma unroll
    for (int t = 0; t < 8; ++t) acc[t] = (v8f){0, 0, 0, 0, 0, 0, 0, 0};

    stage_compress(Y, WcmpT, smem, smem + 2 * K2_SA, m0, n0, 0, tid);

    const int NS = CDIM / 32;   // 64 K-steps
    for (int s = 0; s < NS; ++s) {
        const int cur = s & 1;
        char* sAcur = smem + (size_t)cur * K2_SA;
        char* sBcur = smem + 2 * K2_SA + (size_t)cur * K2_SA;
        if (s + 1 < NS) {
            char* sAnxt = smem + (size_t)(cur ^ 1) * K2_SA;
            char* sBnxt = smem + 2 * K2_SA + (size_t)(cur ^ 1) * K2_SA;
            stage_compress(Y, WcmpT, sAnxt, sBnxt, m0, n0, (s + 1) * 32, tid);
            asm volatile("s_wait_asynccnt 4" ::: "memory");
        } else {
            asm volatile("s_wait_asynccnt 0" ::: "memory");
        }
        __syncthreads();
        v16bf bf = load_frag_b((const unsigned short*)sBcur, wid, lane);
#pragma unroll
        for (int t = 0; t < 8; ++t) {
            v16bf af = load_frag_a((const unsigned short*)sAcur, t, lane);
            acc[t] = __builtin_amdgcn_wmma_f32_16x16x32_bf16(
                false, af, false, bf, (short)0, acc[t], false, false);
        }
        __syncthreads();
    }

    // ---- epilogue: + b_cmp, store f32 ----
    {
        const int nloc  = wid * 16 + (lane & 15);
        const int n     = n0 + nloc;
        const float bias = bcmp[n];
        const int mbase = (lane >> 4) * 8;
#pragma unroll
        for (int t = 0; t < 8; ++t) {
#pragma unroll
            for (int v = 0; v < 8; ++v) {
                out[(size_t)(m0 + t * 16 + mbase + v) * DIM + n] = acc[t][v] + bias;
            }
        }
    }
}

extern "C" void kernel_launch(void* const* d_in, const int* in_sizes, int n_in,
                              void* d_out, int out_size, void* d_ws, size_t ws_size,
                              hipStream_t stream) {
    const float* x     = (const float*)d_in[0];
    const float* Wexp  = (const float*)d_in[1];
    const float* bexp  = (const float*)d_in[2];
    const float* convw = (const float*)d_in[3];
    const float* convb = (const float*)d_in[4];
    const float* Wcmp  = (const float*)d_in[5];
    const float* bcmp  = (const float*)d_in[6];
    float* out = (float*)d_out;

    // Workspace layout (bytes): Y[64MB] | Xbf[32MB] | WexpT[4MB] | WcmpT[4MB]
    char* ws = (char*)d_ws;
    unsigned short* Y     = (unsigned short*)(ws);
    unsigned short* Xbf   = (unsigned short*)(ws + (size_t)MTOT * CDIM * 2);
    unsigned short* WexpT = (unsigned short*)(ws + (size_t)MTOT * CDIM * 2 + (size_t)MTOT * DIM * 2);
    unsigned short* WcmpT = (unsigned short*)(ws + (size_t)MTOT * CDIM * 2 + (size_t)MTOT * DIM * 2
                                                 + (size_t)DIM * CDIM * 2);

    cvt_x_kernel<<<(MTOT * DIM) / 1024, 256, 0, stream>>>(x, Xbf);
    transpose_cvt_kernel<<<dim3(DIM / 32, CDIM / 32), 256, 0, stream>>>(Wexp, WexpT, DIM, CDIM);
    transpose_cvt_kernel<<<dim3(CDIM / 32, DIM / 32), 256, 0, stream>>>(Wcmp, WcmpT, CDIM, DIM);

    expand_conv_kernel<<<dim3(MTOT / 128, CDIM / 128), 256, 0, stream>>>(
        Xbf, WexpT, bexp, convw, convb, Y);
    compress_kernel<<<dim3(MTOT / 128, DIM / 128), 256, 0, stream>>>(Y, WcmpT, bcmp, out);
}
